// MultiScaleAttention_16492674417080
// MI455X (gfx1250) — compile-verified, matrix operand order checked by reading, not run on hardware
//
#include <hip/hip_runtime.h>
#include <hip/hip_bf16.h>

typedef __bf16 bf16;
typedef __attribute__((ext_vector_type(16))) __bf16 bf16x16;
typedef __attribute__((ext_vector_type(8)))  float  f32x8;

__device__ __forceinline__ bf16 f2bf(float f) {
  unsigned u = __builtin_bit_cast(unsigned, f);
  unsigned r = (u + 0x7FFFu + ((u >> 16) & 1u)) >> 16;
  unsigned short s = (unsigned short)r;
  return __builtin_bit_cast(bf16, s);
}
__device__ __forceinline__ float bf2f(bf16 b) {
  unsigned short s = __builtin_bit_cast(unsigned short, b);
  unsigned u = ((unsigned)s) << 16;
  return __builtin_bit_cast(float, u);
}

union FragU { bf16x16 v; uint4 u[2]; };

__device__ __forceinline__ f32x8 wmma_bf16(bf16x16 a, bf16x16 b, f32x8 c) {
  return __builtin_amdgcn_wmma_f32_16x16x32_bf16(false, a, false, b, (short)0, c,
                                                 false, false);
}

// Load a 16x32 bf16 A-style fragment: lane m=lane&15 holds row m, K-half per lane>>4,
// elements 0..7 at K=kh..kh+7, 8..15 at K=kh+16..kh+23 (two contiguous 16B chunks).
__device__ __forceinline__ bf16x16 load_frag(const bf16* base, int row_stride, int lane) {
  const bf16* p = base + (lane & 15) * row_stride + ((lane >> 4) << 3);
  FragU t;
  t.u[0] = *(const uint4*)(p);
  t.u[1] = *(const uint4*)(p + 16);
  return t.v;
}

// ---------------- cast f32 -> bf16 ----------------
__global__ void k_cast_bf16(const float* __restrict__ in, bf16* __restrict__ out, int n) {
  int i = (blockIdx.x * blockDim.x + threadIdx.x) * 4;
  if (i + 3 < n) {
    float4 v = *(const float4*)(in + i);
    out[i] = f2bf(v.x); out[i + 1] = f2bf(v.y);
    out[i + 2] = f2bf(v.z); out[i + 3] = f2bf(v.w);
  }
}

// ---------------- pack weights into B-fragment order ----------------
// Packed layout: [globalNtile][ktile][lane 0..31][elem 0..15] bf16
__global__ void k_pack_w(const float* __restrict__ W, bf16* __restrict__ P,
                         int K, int Ncols, int ntOffset) {
  int KT = K >> 5;
  long idx = (long)blockIdx.x * blockDim.x + threadIdx.x;
  long total = (long)(Ncols >> 4) * KT * 512;
  if (idx >= total) return;
  int e = idx & 15;
  int lane = (idx >> 4) & 31;
  long rem = idx >> 9;
  int kt = (int)(rem % KT);
  int ntl = (int)(rem / KT);
  int n = (ntl << 4) + (lane & 15);
  int k = (kt << 5) + ((lane >> 4) << 3) + (e < 8 ? e : e + 8);
  P[((long)(ntl + ntOffset) * KT + kt) * 512 + lane * 16 + e] =
      f2bf(W[(long)k * Ncols + n]);
}

// ---------------- WMMA GEMM core: 2x2 tiles (32x32) per wave ----------------
// Block: 256 threads = 8 waves as 4(M) x 2(N); block tile 128 x 64.
// Per k-step: 4 fragment loads feed 4 WMMAs (each fragment reused twice).
struct Acc4 { f32x8 c00, c01, c10, c11; };

__device__ __forceinline__ Acc4 gemm_core(const bf16* __restrict__ A,
                                          const bf16* __restrict__ Bp,
                                          int K, int mBase, int ntBase, int lane) {
  int KT = K >> 5;
  Acc4 acc;
  acc.c00 = f32x8{0.f, 0.f, 0.f, 0.f, 0.f, 0.f, 0.f, 0.f};
  acc.c01 = acc.c00; acc.c10 = acc.c00; acc.c11 = acc.c00;
  const bf16* arow0 = A + (size_t)(mBase + (lane & 15)) * K + ((lane >> 4) << 3);
  const bf16* arow1 = arow0 + (size_t)16 * K;
  const bf16* bf0 = Bp + (size_t)ntBase * KT * 512 + lane * 16;
  const bf16* bf1 = bf0 + (size_t)KT * 512;
  for (int kt = 0; kt < KT; ++kt) {
    FragU a0, a1, b0, b1;
    a0.u[0] = *(const uint4*)(arow0);
    a0.u[1] = *(const uint4*)(arow0 + 16);
    a1.u[0] = *(const uint4*)(arow1);
    a1.u[1] = *(const uint4*)(arow1 + 16);
    b0.u[0] = *(const uint4*)(bf0);
    b0.u[1] = *(const uint4*)(bf0 + 8);
    b1.u[0] = *(const uint4*)(bf1);
    b1.u[1] = *(const uint4*)(bf1 + 8);
    __builtin_prefetch((const void*)(arow0 + 256), 0, 1);  // stream-ahead on A
    acc.c00 = wmma_bf16(a0.v, b0.v, acc.c00);
    acc.c01 = wmma_bf16(a0.v, b1.v, acc.c01);
    acc.c10 = wmma_bf16(a1.v, b0.v, acc.c10);
    acc.c11 = wmma_bf16(a1.v, b1.v, acc.c11);
    arow0 += 32; arow1 += 32; bf0 += 512; bf1 += 512;
  }
  return acc;
}

// bf16 output (QKV projection)
__global__ __launch_bounds__(256) void k_gemm_bf16(
    const bf16* __restrict__ A, const bf16* __restrict__ Bp,
    bf16* __restrict__ Cb, int K, int Ncols) {
  int lane = threadIdx.x & 31;
  int wave = threadIdx.x >> 5;
  int wm = wave & 3, wn = wave >> 2;
  int mBase = blockIdx.x * 128 + wm * 32;
  int ntBase = blockIdx.y * 4 + wn * 2;
  Acc4 acc = gemm_core(A, Bp, K, mBase, ntBase, lane);
  int n0 = (ntBase << 4) + (lane & 15);
  int mh = (lane >> 4) << 3;
  bf16* p0 = Cb + (size_t)(mBase + mh) * Ncols + n0;        // rows mBase..+16
  bf16* p1 = Cb + (size_t)(mBase + 16 + mh) * Ncols + n0;   // rows mBase+16..+32
  for (int r = 0; r < 8; ++r) {
    size_t ro = (size_t)r * Ncols;
    p0[ro]      = f2bf(acc.c00[r]);
    p0[ro + 16] = f2bf(acc.c01[r]);
    p1[ro]      = f2bf(acc.c10[r]);
    p1[ro + 16] = f2bf(acc.c11[r]);
  }
}

// f32 output + bias (final projection)
__global__ __launch_bounds__(256) void k_gemm_f32(
    const bf16* __restrict__ A, const bf16* __restrict__ Bp,
    const float* __restrict__ bias, float* __restrict__ Out, int K, int Ncols) {
  int lane = threadIdx.x & 31;
  int wave = threadIdx.x >> 5;
  int wm = wave & 3, wn = wave >> 2;
  int mBase = blockIdx.x * 128 + wm * 32;
  int ntBase = blockIdx.y * 4 + wn * 2;
  Acc4 acc = gemm_core(A, Bp, K, mBase, ntBase, lane);
  int n0 = (ntBase << 4) + (lane & 15);
  int mh = (lane >> 4) << 3;
  float b0 = bias[n0], b1 = bias[n0 + 16];
  float* p0 = Out + (size_t)(mBase + mh) * Ncols + n0;
  float* p1 = Out + (size_t)(mBase + 16 + mh) * Ncols + n0;
  for (int r = 0; r < 8; ++r) {
    size_t ro = (size_t)r * Ncols;
    p0[ro]      = acc.c00[r] + b0;
    p0[ro + 16] = acc.c01[r] + b1;
    p1[ro]      = acc.c10[r] + b0;
    p1[ro + 16] = acc.c11[r] + b1;
  }
}

// ---------------- depthwise 2x2 conv (q,k,v together) ----------------
__global__ void k_dwconv(const bf16* __restrict__ qkv, const float* __restrict__ cw,
                         const float* __restrict__ cb,
                         bf16* __restrict__ oq, bf16* __restrict__ ok2,
                         bf16* __restrict__ ov,
                         int headBase, int Ho, int Wo, int stride, int pad, int dil) {
  long idx = (long)blockIdx.x * blockDim.x + threadIdx.x;
  long total = (long)32 * 4 * Ho * Wo * 32;
  if (idx >= total) return;
  int ch = idx & 31;
  long t = idx >> 5;
  int xo = (int)(t % Wo); t /= Wo;
  int yo = (int)(t % Ho); t /= Ho;
  int hg = (int)(t & 3);
  int b  = (int)(t >> 2);
  float accq = cb[ch], acck = cb[ch], accv = cb[ch];
  int col = (headBase + hg) * 32 + ch;
  for (int ky = 0; ky < 2; ++ky) {
    int iy = yo * stride - pad + ky * dil;
    if (iy < 0 || iy >= 56) continue;
    for (int kx = 0; kx < 2; ++kx) {
      int ix = xo * stride - pad + kx * dil;
      if (ix < 0 || ix >= 56) continue;
      float w = cw[(ky * 2 + kx) * 32 + ch];
      size_t base = ((size_t)b * 3136 + (size_t)iy * 56 + ix) * 1152 + col;
      accq += w * bf2f(qkv[base]);
      acck += w * bf2f(qkv[base + 384]);
      accv += w * bf2f(qkv[base + 768]);
    }
  }
  oq[idx] = f2bf(accq);
  ok2[idx] = f2bf(acck);
  ov[idx] = f2bf(accv);
}

// ---------------- windowed attention (49 tokens, d=32) ----------------
__global__ __launch_bounds__(128) void k_attn(
    const bf16* __restrict__ Qsrc, const bf16* __restrict__ Ksrc,
    const bf16* __restrict__ Vsrc,
    long strideB, long strideH, long strideY, long strideX,
    int gridW, int nwin, int reps, int headBase,
    bf16* __restrict__ ybuf) {
  __shared__ bf16 Qs[64 * 32];
  __shared__ bf16 Ks[64 * 32];
  __shared__ bf16 Vt[32 * 64];   // transposed: [ch][tok]
  __shared__ bf16 Ps[64 * 64];   // softmax probs (bf16, padded)
  int tid = threadIdx.x;
  int lane = tid & 31;
  int wave = tid >> 5;
  int w = blockIdx.x % nwin;
  int hg = (blockIdx.x / nwin) & 3;
  int b = blockIdx.x / (nwin * 4);
  int wr = w / gridW, wc = w % gridW;
  long base = (long)b * strideB + (long)hg * strideH;

  for (int e = tid; e < 2048; e += 128) {
    int t = e >> 5, ch = e & 31;
    bf16 qv = f2bf(0.f), kv = qv, vv = qv;
    if (t < 49) {
      int ty = t / 7, tx = t % 7;
      long a = base + (long)(wr * 7 + ty) * strideY + (long)(wc * 7 + tx) * strideX + ch;
      qv = Qsrc[a]; kv = Ksrc[a]; vv = Vsrc[a];
    }
    Qs[t * 32 + ch] = qv;
    Ks[t * 32 + ch] = kv;
    Vt[ch * 64 + t] = vv;
  }
  __syncthreads();

  // S = Q K^T : wave owns 16-row tile mi = wave
  f32x8 acc[4];
  {
    bf16x16 a = load_frag(Qs + wave * 16 * 32, 32, lane);
    for (int ni = 0; ni < 4; ++ni) {
      bf16x16 bb = load_frag(Ks + ni * 16 * 32, 32, lane);
      f32x8 z = {0.f, 0.f, 0.f, 0.f, 0.f, 0.f, 0.f, 0.f};
      acc[ni] = wmma_bf16(a, bb, z);
    }
  }

  // register softmax: row m = wave*16 + 8*(lane>>4) + r spread over 16 lanes x 4 regs
  const float scale = 0.17677669529663687f;  // 32^-0.5
  int ncol = lane & 15;
  int mh = (lane >> 4) << 3;
  for (int r = 0; r < 8; ++r) {
    float v[4];
    float mx = -1e30f;
    for (int ni = 0; ni < 4; ++ni) {
      float s = acc[ni][r] * scale;
      if (ni * 16 + ncol >= 49) s = -1e30f;
      v[ni] = s;
      mx = fmaxf(mx, s);
    }
    for (int off = 1; off < 16; off <<= 1) mx = fmaxf(mx, __shfl_xor(mx, off, 32));
    float sum = 0.f;
    for (int ni = 0; ni < 4; ++ni) { v[ni] = __expf(v[ni] - mx); sum += v[ni]; }
    for (int off = 1; off < 16; off <<= 1) sum += __shfl_xor(sum, off, 32);
    float inv = 1.f / sum;
    int m = wave * 16 + mh + r;
    for (int ni = 0; ni < 4; ++ni)
      Ps[m * 64 + ni * 16 + ncol] = f2bf(v[ni] * inv);
  }
  __syncthreads();

  // Y = P V : wave owns rows mi=wave, two N-tiles (ch 0..15 / 16..31), K=64
  f32x8 y0 = {0.f, 0.f, 0.f, 0.f, 0.f, 0.f, 0.f, 0.f};
  f32x8 y1 = y0;
  for (int kt = 0; kt < 2; ++kt) {
    bf16x16 a  = load_frag(Ps + wave * 16 * 64 + kt * 32, 64, lane);
    bf16x16 b0 = load_frag(Vt + kt * 32, 64, lane);
    bf16x16 b1 = load_frag(Vt + 16 * 64 + kt * 32, 64, lane);
    y0 = wmma_bf16(a, b0, y0);
    y1 = wmma_bf16(a, b1, y1);
  }

  int head = headBase + hg;
  for (int r = 0; r < 8; ++r) {
    int m = wave * 16 + mh + r;
    if (m >= 49) continue;
    bf16 o0 = f2bf(y0[r]), o1 = f2bf(y1[r]);
    for (int rep = 0; rep < reps; ++rep) {
      int region = w + rep * nwin;   // jnp.tile: full region r maps to window r % nwin
      size_t tok = (size_t)b * 3136 + (size_t)region * 49 + m;
      bf16* op = ybuf + tok * 384 + head * 32;
      op[ncol] = o0;
      op[16 + ncol] = o1;
    }
  }
}

extern "C" void kernel_launch(void* const* d_in, const int* in_sizes, int n_in,
                              void* d_out, int out_size, void* d_ws, size_t ws_size,
                              hipStream_t stream) {
  const float* x      = (const float*)d_in[0];
  const float* Wq     = (const float*)d_in[1];
  const float* Wkv    = (const float*)d_in[2];
  const float* Wproj  = (const float*)d_in[3];
  const float* bproj  = (const float*)d_in[4];
  const float* c1w    = (const float*)d_in[5];
  const float* c1b    = (const float*)d_in[6];
  const float* c2w    = (const float*)d_in[7];
  const float* c2b    = (const float*)d_in[8];

  const int M = 100352;           // B*N = 32*3136
  const int C = 384;
  const int nXC = M * C;          // 38,535,168

  char* base = (char*)d_ws;
  size_t off = 0;
  auto take = [&](size_t bytes) -> void* {
    void* p = base + off;
    off += (bytes + 255) & ~(size_t)255;
    return p;
  };
  bf16* xb    = (bf16*)take((size_t)nXC * 2);
  bf16* qkv   = (bf16*)take((size_t)M * 1152 * 2);
  bf16* pQKV  = (bf16*)take((size_t)72 * 12 * 512 * 2);
  bf16* pProj = (bf16*)take((size_t)24 * 12 * 512 * 2);
  bf16* qs1   = (bf16*)take((size_t)3211264 * 2);
  bf16* ks1   = (bf16*)take((size_t)3211264 * 2);
  bf16* vs1   = (bf16*)take((size_t)3211264 * 2);
  bf16* qs2   = (bf16*)take((size_t)802816 * 2);
  bf16* ks2   = (bf16*)take((size_t)802816 * 2);
  bf16* vs2   = (bf16*)take((size_t)802816 * 2);
  bf16* ybuf  = (bf16*)take((size_t)nXC * 2);
  (void)ws_size;

  // 1) cast activations to bf16
  k_cast_bf16<<<dim3(nXC / 1024), 256, 0, stream>>>(x, xb, nXC);

  // 2) pack weights into fragment layout (Wq cols 0..383, Wkv cols 384..1151)
  k_pack_w<<<dim3(576),  256, 0, stream>>>(Wq,    pQKV,  384, 384,  0);
  k_pack_w<<<dim3(1152), 256, 0, stream>>>(Wkv,   pQKV,  384, 768, 24);
  k_pack_w<<<dim3(576),  256, 0, stream>>>(Wproj, pProj, 384, 384,  0);

  // 3) fused QKV projection: qkv[M,1152] = xb @ [Wq|Wkv]
  k_gemm_bf16<<<dim3(M / 128, 18), 256, 0, stream>>>(xb, pQKV, qkv, 384, 1152);

  // 4) depthwise downsampling for scales 1 and 2
  k_dwconv<<<dim3(12544), 256, 0, stream>>>(qkv, c1w, c1b, qs1, ks1, vs1,
                                            4, 28, 28, 2, 0, 1);
  k_dwconv<<<dim3(3136),  256, 0, stream>>>(qkv, c2w, c2b, qs2, ks2, vs2,
                                            8, 14, 14, 4, 1, 2);

  // 5) windowed attention per scale -> ybuf [B, N(window-major), C] bf16
  // scale 0: full res, source = qkv (q at +0, k at +384, v at +768)
  k_attn<<<dim3(8192), 128, 0, stream>>>(
      qkv, qkv + 384, qkv + 768,
      (long)3136 * 1152, 32L, (long)56 * 1152, 1152L,
      8, 64, 1, 0, ybuf);
  // scale 1: 28x28
  k_attn<<<dim3(2048), 128, 0, stream>>>(
      qs1, ks1, vs1,
      (long)4 * 28 * 28 * 32, (long)28 * 28 * 32, (long)28 * 32, 32L,
      4, 16, 4, 4, ybuf);
  // scale 2: 14x14
  k_attn<<<dim3(512), 128, 0, stream>>>(
      qs2, ks2, vs2,
      (long)4 * 14 * 14 * 32, (long)14 * 14 * 32, (long)14 * 32, 32L,
      2, 4, 16, 8, ybuf);

  // 6) output projection: out[M,384] = ybuf @ Wproj + bproj (f32)
  k_gemm_f32<<<dim3(M / 128, 6), 256, 0, stream>>>(ybuf, pProj, bproj,
                                                   (float*)d_out, 384, 384);
}
